// MultiHeadSelfAttention_80719615361347
// MI455X (gfx1250) — matrix operand order repaired and checked by reading, not compiled, not run on hardware
//
#include <hip/hip_runtime.h>

// ---------------------------------------------------------------------------
// MI455X (gfx1250) multi-head self-attention forward.
// bf16 WMMA (v_wmma_f32_16x16x32_bf16) for all matmuls, fp32 accumulate,
// flash-attention streaming softmax, async global->LDS staging (ASYNCcnt).
//   x:(4,2048,1024) f32 ; Wq/Wk/Wv/Wo:(1024,1024) f32 ; b*:(1024,) f32
//   out:(4,2048,1024) f32
// ---------------------------------------------------------------------------

typedef __bf16 bf16_t;
typedef __attribute__((ext_vector_type(16))) __bf16        v16bf;
typedef __attribute__((ext_vector_type(8)))  float         v8f;
typedef __attribute__((ext_vector_type(4)))  unsigned int  u32x4;
typedef __attribute__((ext_vector_type(4)))  int           v4i;

#define B_NUM  4
#define S_LEN  2048
#define D_DIM  1024
#define H_NUM  16
#define HD     64
#define M_ROWS (B_NUM * S_LEN)   // 8192

#if defined(__has_builtin)
# if __has_builtin(__builtin_amdgcn_global_load_async_to_lds_b128)
#  define HAVE_ASYNC_LDS 1
# endif
#endif
#ifndef HAVE_ASYNC_LDS
# define HAVE_ASYNC_LDS 0
#endif

typedef __attribute__((address_space(1))) v4i* gv4i_ptr;   // global int4*
typedef __attribute__((address_space(3))) v4i* lv4i_ptr;   // LDS    int4*

union Frag16 {
  v16bf        v;
  u32x4        q[2];
  unsigned int u[8];
};

static __device__ inline v8f wmma_bf16(v16bf a, v16bf b, v8f c) {
  return __builtin_amdgcn_wmma_f32_16x16x32_bf16(
      false, a, false, b, (short)0, c, false, false);
}

// Async DMA of one 16-byte chunk global -> LDS (no VGPR round trip, ASYNCcnt).
static __device__ inline void async_copy_b128(const bf16_t* gsrc, bf16_t* ldst) {
#if HAVE_ASYNC_LDS
  __builtin_amdgcn_global_load_async_to_lds_b128(
      (gv4i_ptr)gsrc, (lv4i_ptr)ldst, /*imm offset*/0, /*cpol*/0);
#else
  *(u32x4*)ldst = *(const u32x4*)gsrc;   // fallback: synchronous copy
#endif
}

static __device__ inline void wait_async_all() {
#if HAVE_ASYNC_LDS
# if __has_builtin(__builtin_amdgcn_s_wait_asynccnt)
  __builtin_amdgcn_s_wait_asynccnt(0);
# else
  asm volatile("s_wait_asynccnt 0" ::: "memory");
# endif
#endif
}

// A fragment (16 rows x 32 K), row-major, K contiguous -> two b128 loads.
// Lane r=lane&15 holds row r; VGPR v: K = (v<4 ? 2v : 16+2(v-4)) + 8*(lane>>4).
static __device__ inline v16bf load_afrag(const bf16_t* base, int ld, int row0, int k0) {
  const int lane = threadIdx.x & 31;
  const int r    = lane & 15;
  const int hh   = lane >> 4;
  const bf16_t* p = base + (size_t)(row0 + r) * ld + k0 + 8 * hh;
  Frag16 f;
  f.q[0] = *(const u32x4*)(p);
  f.q[1] = *(const u32x4*)(p + 16);
  return f.v;
}

// B fragment (32 K x 16 cols) from column-K-contiguous storage -> two b128.
// Lane n holds column n; VGPR v: K = (v<4 ? 2v : 16+2(v-4)) + 8*(lane>>4).
static __device__ inline v16bf load_bfrag(const bf16_t* base, int ld, int col0, int k0) {
  const int lane = threadIdx.x & 31;
  const int n    = lane & 15;
  const int hh   = lane >> 4;
  const bf16_t* p = base + (size_t)(col0 + n) * ld + k0 + 8 * hh;
  Frag16 f;
  f.q[0] = *(const u32x4*)(p);
  f.q[1] = *(const u32x4*)(p + 16);
  return f.v;
}

// A fragment from LDS with padded rows (dword loads; 4-byte aligned always).
static __device__ inline v16bf load_afrag_lds(const bf16_t* base, int ld) {
  const int lane = threadIdx.x & 31;
  const int r    = lane & 15;
  const int hh   = lane >> 4;
  const unsigned int* p = (const unsigned int*)(base + (size_t)r * ld);
  Frag16 f;
#pragma unroll
  for (int v = 0; v < 8; ++v) {
    const int kk = ((v < 4) ? 2 * v : 16 + 2 * (v - 4)) + 8 * hh;
    f.u[v] = p[kk >> 1];
  }
  return f.v;
}

// ---------------------------------------------------------------------------
// fp32 -> bf16 conversion kernels
// ---------------------------------------------------------------------------
__global__ void cvt_f32_bf16(const float* __restrict__ in, bf16_t* __restrict__ out, int n) {
  int i = blockIdx.x * 256 + threadIdx.x;
  if (i < n) out[i] = (bf16_t)in[i];
}

// WT[n][k] = (bf16) W[k][n]   (D x D)
__global__ void transpose_w_bf16(const float* __restrict__ W, bf16_t* __restrict__ WT) {
  int n = blockIdx.x * 256 + threadIdx.x;
  int k = blockIdx.y;
  WT[(size_t)n * D_DIM + k] = (bf16_t)W[(size_t)k * D_DIM + n];
}

// ---------------------------------------------------------------------------
// GEMM: Y[M,N] = Xbf[M,K] @ WT[N,K]^T + bias
//   mode 0: fp32 row-major output (final projection)
//   mode 1: bf16 output, head-split (B,H,S,Hd)            (Q, K)
//   mode 2: bf16 output, head-split transposed (B,H,Hd,S) (V)
// 256 threads = 8 waves, 4(M) x 2(N); wave tile 32x64; block tile 128x128.
// Weight tile (128 cols x 32 K) staged in LDS via async DMA, double-buffered.
// ---------------------------------------------------------------------------
__global__ __launch_bounds__(256)
void gemm_bf16_wmma(const bf16_t* __restrict__ X,
                    const bf16_t* __restrict__ WT,
                    const float*  __restrict__ bias,
                    float*        __restrict__ outF,
                    bf16_t*       __restrict__ outB,
                    int Msz, int Nsz, int Ksz, int mode) {
  // pad 40: row stride 80 B = 5x16B (b128-aligned), 20-dword bank stride
  __shared__ __align__(16) bf16_t ldsB[2][128][40];

  const int wave = threadIdx.x >> 5;
  const int lane = threadIdx.x & 31;
  const int wm   = wave & 3;
  const int wn   = wave >> 2;
  const int row0     = blockIdx.y * 128 + wm * 32;
  const int colBlock = blockIdx.x * 128;
  const int col0     = colBlock + wn * 64;
  const int t = threadIdx.x;

  v8f acc[2][4];
  const v8f vzero = {0.f, 0.f, 0.f, 0.f, 0.f, 0.f, 0.f, 0.f};
#pragma unroll
  for (int i = 0; i < 2; ++i)
#pragma unroll
    for (int j = 0; j < 4; ++j) acc[i][j] = vzero;

  // stage one 128x32 weight tile: 512 x 16B chunks, 2 per thread
  auto stage = [&](int bufi, int k0) {
    {
      const int c = t, row = c >> 2, q = c & 3;
      async_copy_b128(WT + (size_t)(colBlock + row) * Ksz + k0 + q * 8,
                      &ldsB[bufi][row][q * 8]);
    }
    {
      const int c = t + 256, row = c >> 2, q = c & 3;
      async_copy_b128(WT + (size_t)(colBlock + row) * Ksz + k0 + q * 8,
                      &ldsB[bufi][row][q * 8]);
    }
  };

  stage(0, 0);
  int buf = 0;
  for (int k0 = 0; k0 < Ksz; k0 += 32) {
    wait_async_all();
    __syncthreads();
    if (k0 + 32 < Ksz) stage(buf ^ 1, k0 + 32);

    const v16bf a0 = load_afrag(X, Ksz, row0,      k0);
    const v16bf a1 = load_afrag(X, Ksz, row0 + 16, k0);
    v16bf bfr[4];
#pragma unroll
    for (int j = 0; j < 4; ++j)
      bfr[j] = load_bfrag(&ldsB[buf][0][0], 40, wn * 64 + 16 * j, 0);
#pragma unroll
    for (int j = 0; j < 4; ++j) {
      acc[0][j] = wmma_bf16(a0, bfr[j], acc[0][j]);
      acc[1][j] = wmma_bf16(a1, bfr[j], acc[1][j]);
    }
    __syncthreads();
    buf ^= 1;
  }

  const int nl = lane & 15;
  const int hh = lane >> 4;
#pragma unroll
  for (int i = 0; i < 2; ++i)
#pragma unroll
    for (int j = 0; j < 4; ++j)
#pragma unroll
      for (int v = 0; v < 8; ++v) {
        const int m = row0 + 16 * i + 8 * hh + v;
        const int n = col0 + 16 * j + nl;
        const float val = acc[i][j][v] + bias[n];
        if (mode == 0) {
          outF[(size_t)m * Nsz + n] = val;
        } else {
          const int b = m >> 11;
          const int s = m & (S_LEN - 1);
          const int hd = n >> 6;
          const int d  = n & (HD - 1);
          if (mode == 1)
            outB[(((size_t)(b * H_NUM + hd)) * S_LEN + s) * HD + d] = (bf16_t)val;
          else
            outB[(((size_t)(b * H_NUM + hd)) * HD + d) * S_LEN + s] = (bf16_t)val;
        }
      }
}

// ---------------------------------------------------------------------------
// Flash attention: one (b,h) x 64-query block per workgroup (4 waves, wave32),
// 16-query tile per wave, streaming K/V in 32-key steps.
// K/V tiles staged once per step into LDS via async DMA (shared by all waves).
//   Q,K : (B,H,S,Hd) bf16     V : (B,H,Hd,S) bf16     O : (B,S,D) bf16
// ---------------------------------------------------------------------------
__global__ __launch_bounds__(128)
void attn_wmma(const bf16_t* __restrict__ Q,
               const bf16_t* __restrict__ Kt,
               const bf16_t* __restrict__ Vt,
               bf16_t*       __restrict__ O) {
  __shared__ __align__(16) bf16_t ldsK[2][32][72];  // 32 keys x 64 hd (pad 72)
  __shared__ __align__(16) bf16_t ldsV[2][64][40];  // 64 hd  x 32 keys (pad 40)
  __shared__ __align__(16) bf16_t ldsP[4][16][40];  // per-wave P relayout tile

  const int wave = threadIdx.x >> 5;
  const int lane = threadIdx.x & 31;
  const int t    = threadIdx.x;
  const int b    = blockIdx.z;
  const int head = blockIdx.y;
  const size_t bh = (size_t)b * H_NUM + head;
  const bf16_t* Qb = Q  + bh * S_LEN * HD;
  const bf16_t* Kb = Kt + bh * S_LEN * HD;
  const bf16_t* Vb = Vt + bh * HD * S_LEN;
  const int q0 = blockIdx.x * 64 + wave * 16;
  const int nl = lane & 15;
  const int hh = lane >> 4;

  const v16bf qf0 = load_afrag(Qb, HD, q0, 0);
  const v16bf qf1 = load_afrag(Qb, HD, q0, 32);

  const v8f vzero = {0.f, 0.f, 0.f, 0.f, 0.f, 0.f, 0.f, 0.f};
  v8f o_acc[4] = {vzero, vzero, vzero, vzero};
  float m_i[8], l_i[8];
#pragma unroll
  for (int v = 0; v < 8; ++v) { m_i[v] = -3.0e38f; l_i[v] = 0.f; }

  const float scale = 0.125f;   // 1/sqrt(64)

  // stage 32x64 K tile (256 chunks) + 64x32 V tile (256 chunks); 4 chunks/thread
  auto stage_kv = [&](int bufi, int kt2) {
#pragma unroll
    for (int it = 0; it < 2; ++it) {
      const int c = t + it * 128;
      const int kr = c >> 3, kq = c & 7;
      async_copy_b128(Kb + (size_t)(kt2 + kr) * HD + kq * 8,
                      &ldsK[bufi][kr][kq * 8]);
    }
#pragma unroll
    for (int it = 0; it < 2; ++it) {
      const int c = t + it * 128;
      const int vr = c >> 2, vq = c & 3;
      async_copy_b128(Vb + (size_t)vr * S_LEN + kt2 + vq * 8,
                      &ldsV[bufi][vr][vq * 8]);
    }
  };

  stage_kv(0, 0);
  int buf = 0;
  for (int kt = 0; kt < S_LEN; kt += 32) {
    wait_async_all();
    __syncthreads();
    if (kt + 32 < S_LEN) stage_kv(buf ^ 1, kt + 32);

    // scores: 16x32 tile = two 16x16 WMMA tiles, each over Hd=64 (2 K-steps)
    v8f sc0 = vzero, sc1 = vzero;
    sc0 = wmma_bf16(qf0, load_bfrag(&ldsK[buf][0][0], 72, 0,  0),  sc0);
    sc0 = wmma_bf16(qf1, load_bfrag(&ldsK[buf][0][0], 72, 0,  32), sc0);
    sc1 = wmma_bf16(qf0, load_bfrag(&ldsK[buf][0][0], 72, 16, 0),  sc1);
    sc1 = wmma_bf16(qf1, load_bfrag(&ldsK[buf][0][0], 72, 16, 32), sc1);

    // online softmax; each 16-lane half owns rows {8*hh + v}
#pragma unroll
    for (int v = 0; v < 8; ++v) {
      const float x0 = sc0[v] * scale;
      const float x1 = sc1[v] * scale;
      float tmx = fmaxf(x0, x1);
#pragma unroll
      for (int off = 8; off >= 1; off >>= 1)
        tmx = fmaxf(tmx, __shfl_xor(tmx, off, 16));
      const float mnew  = fmaxf(m_i[v], tmx);
      const float alpha = __expf(m_i[v] - mnew);
      const float p0 = __expf(x0 - mnew);
      const float p1 = __expf(x1 - mnew);
      float rs = p0 + p1;
#pragma unroll
      for (int off = 8; off >= 1; off >>= 1)
        rs += __shfl_xor(rs, off, 16);
      l_i[v] = l_i[v] * alpha + rs;
      m_i[v] = mnew;
#pragma unroll
      for (int t4 = 0; t4 < 4; ++t4) o_acc[t4][v] *= alpha;

      const int prow = 8 * hh + v;
      ldsP[wave][prow][nl]      = (bf16_t)p0;
      ldsP[wave][prow][16 + nl] = (bf16_t)p1;
    }

    // same-wave DS RAW: drain our LDS stores, then reread P as an A fragment
    asm volatile("s_wait_dscnt 0" ::: "memory");
    const v16bf pf = load_afrag_lds(&ldsP[wave][0][0], 40);

    // O += P(16x32) @ V(32x64)
#pragma unroll
    for (int t4 = 0; t4 < 4; ++t4) {
      const v16bf vb = load_bfrag(&ldsV[buf][0][0], 40, 16 * t4, 0);
      o_acc[t4] = wmma_bf16(pf, vb, o_acc[t4]);
    }
    __syncthreads();
    buf ^= 1;
  }

  // normalize, re-merge heads: O[b][s][head*64 + d]
#pragma unroll
  for (int t4 = 0; t4 < 4; ++t4)
#pragma unroll
    for (int v = 0; v < 8; ++v) {
      const int m = q0 + 8 * hh + v;
      const int d = head * HD + 16 * t4 + nl;
      O[((size_t)b * S_LEN + m) * D_DIM + d] = (bf16_t)(o_acc[t4][v] / l_i[v]);
    }
}

// ---------------------------------------------------------------------------
// Launch pipeline
// ---------------------------------------------------------------------------
extern "C" void kernel_launch(void* const* d_in, const int* in_sizes, int n_in,
                              void* d_out, int out_size, void* d_ws, size_t ws_size,
                              hipStream_t stream) {
  const float* x  = (const float*)d_in[0];
  const float* Wq = (const float*)d_in[1];
  const float* bq = (const float*)d_in[2];
  const float* Wk = (const float*)d_in[3];
  const float* bk = (const float*)d_in[4];
  const float* Wv = (const float*)d_in[5];
  const float* bv = (const float*)d_in[6];
  const float* Wo = (const float*)d_in[7];
  const float* bo = (const float*)d_in[8];
  float* out = (float*)d_out;

  char* ws = (char*)d_ws;
  size_t off = 0;
  auto alloc = [&](size_t bytes) -> char* {
    char* p = ws + off;
    off += (bytes + 255) & ~(size_t)255;
    return p;
  };

  const size_t actBytes = (size_t)M_ROWS * D_DIM * sizeof(bf16_t);  // 16 MB
  const size_t wBytes   = (size_t)D_DIM * D_DIM * sizeof(bf16_t);   //  2 MB

  bf16_t* xbf = (bf16_t*)alloc(actBytes);
  bf16_t* WqT = (bf16_t*)alloc(wBytes);
  bf16_t* WkT = (bf16_t*)alloc(wBytes);
  bf16_t* WvT = (bf16_t*)alloc(wBytes);
  bf16_t* WoT = (bf16_t*)alloc(wBytes);
  bf16_t* Qbf = (bf16_t*)alloc(actBytes);   // (B,H,S,Hd)
  bf16_t* Kbf = (bf16_t*)alloc(actBytes);   // (B,H,S,Hd)
  bf16_t* Vbf = (bf16_t*)alloc(actBytes);   // (B,H,Hd,S)
  bf16_t* Obf = (bf16_t*)alloc(actBytes);   // (B,S,D)
  (void)ws_size; (void)in_sizes; (void)n_in; (void)out_size;

  {
    const int n = M_ROWS * D_DIM;
    cvt_f32_bf16<<<(n + 255) / 256, 256, 0, stream>>>(x, xbf, n);
  }
  {
    dim3 tg(D_DIM / 256, D_DIM);
    transpose_w_bf16<<<tg, 256, 0, stream>>>(Wq, WqT);
    transpose_w_bf16<<<tg, 256, 0, stream>>>(Wk, WkT);
    transpose_w_bf16<<<tg, 256, 0, stream>>>(Wv, WvT);
    transpose_w_bf16<<<tg, 256, 0, stream>>>(Wo, WoT);
  }

  dim3 gg(D_DIM / 128, M_ROWS / 128);   // (8, 64)
  gemm_bf16_wmma<<<gg, 256, 0, stream>>>(xbf, WqT, bq, nullptr, Qbf,
                                         M_ROWS, D_DIM, D_DIM, 1);
  gemm_bf16_wmma<<<gg, 256, 0, stream>>>(xbf, WkT, bk, nullptr, Kbf,
                                         M_ROWS, D_DIM, D_DIM, 1);
  gemm_bf16_wmma<<<gg, 256, 0, stream>>>(xbf, WvT, bv, nullptr, Vbf,
                                         M_ROWS, D_DIM, D_DIM, 2);

  dim3 ga(S_LEN / 64, H_NUM, B_NUM);    // (32, 16, 4)
  attn_wmma<<<ga, 128, 0, stream>>>(Qbf, Kbf, Vbf, Obf);

  gemm_bf16_wmma<<<gg, 256, 0, stream>>>(Obf, WoT, bo, out, nullptr,
                                         M_ROWS, D_DIM, D_DIM, 0);
}
